// CausalMambaSA_56281251446877
// MI455X (gfx1250) — matrix operand order repaired and checked by reading, not compiled, not run on hardware
//
#include <hip/hip_runtime.h>

// ---------------------------------------------------------------------------
// CausalMambaSA forward for MI455X (gfx1250, wave32, WMMA).
// GEMM: operands pre-packed to f16 in exact WMMA per-lane fragment order
//       (ISA 7.12.2 wave32 layouts); inner loop = 4 contiguous v16h loads +
//       4x v_wmma_f32_16x16x32_f16. Direct global->VGPR (no LDS/TDM staging:
//       one wave per tile => no cross-wave reuse to exploit).
// Fusions: softplus(+b_dt) and silu(z)-gating folded into the scan;
//       conv and scan emit f16 fragment-packed copies so the W_x / W_out
//       GEMMs need no separate pack pass.
// Scans: one thread per (batch, channel), state in VGPRs (templated N),
//        float4 loads of Bm/Cm in the sequential recurrence.
// softmax over singleton axis => modality weights identically 1 (elided).
// ---------------------------------------------------------------------------

typedef __attribute__((ext_vector_type(16))) _Float16 v16h;
typedef __attribute__((ext_vector_type(8)))  float    v8f;

#define HD 128

// A-fragment packed index for element (m, k): layout [mp][kp][lane][16].
// Inverse of: k%32 = hi*8 + 2j + h + (j>=4 ? 8 : 0), lane = hi*16 + m%16.
__device__ __forceinline__ long ap_index(long m, int k, int KP)
{
    int kr = k & 31;
    int hi = (kr >> 3) & 1;
    int e  = 2 * (((kr & 16) ? 4 : 0) + ((kr & 7) >> 1)) + (kr & 1);
    int lane = hi * 16 + (int)(m & 15);
    return (((m >> 4) * KP + (k >> 5)) * 32 + lane) * 16 + e;
}

// ===================== operand packing (f32 -> f16 fragments) ==============

// A(M x K, row stride lda) -> AP[mp][kp][lane][16] per A-fragment layout.
__global__ void pack_a(const float* __restrict__ A, int lda, long a_off,
                       int M, int K, _Float16* __restrict__ AP, int KP)
{
    long i = (long)blockIdx.x * blockDim.x + threadIdx.x;
    long total = (long)(M / 16) * KP * 512;
    if (i >= total) return;
    int e    = (int)(i & 15);
    int lane = (int)((i >> 4) & 31);
    long pi  = i >> 9;
    int kp = (int)(pi % KP);
    int mp = (int)(pi / KP);
    int j = e >> 1, h = e & 1;
    int hi = lane >> 4;
    int k = kp * 32 + hi * 8 + 2 * j + h + ((j >= 4) ? 8 : 0);
    int m = mp * 16 + (lane & 15);
    float v = (k < K) ? A[a_off + (long)m * lda + k] : 0.0f;
    AP[i] = (_Float16)v;
}

// W(K x N, row stride ldw) -> WP[kp][np][lane][16] per B-fragment layout.
__global__ void pack_w(const float* __restrict__ W, int ldw,
                       int K, int N, _Float16* __restrict__ WP, int KP, int NP)
{
    long i = (long)blockIdx.x * blockDim.x + threadIdx.x;
    long total = (long)KP * NP * 512;
    if (i >= total) return;
    int e    = (int)(i & 15);
    int lane = (int)((i >> 4) & 31);
    long pi  = i >> 9;
    int np = (int)(pi % NP);
    int kp = (int)(pi / NP);
    int j = e >> 1, h = e & 1;
    int k = kp * 32 + (lane >> 4) * 8 + 2 * j + h + ((j >= 4) ? 8 : 0);
    int n = np * 16 + (lane & 15);
    float v = (k < K && n < N) ? W[(long)k * ldw + n] : 0.0f;
    WP[i] = (_Float16)v;
}

// ======================= WMMA tiled GEMM (packed operands) =================
// One wave computes a 32x32 output tile (2x2 grid of 16x16 WMMA tiles).
// REQUIRES: M % 32 == 0 (true for every call in this model).
__global__ __launch_bounds__(32)
void gemm_pk(const _Float16* __restrict__ AP, const _Float16* __restrict__ WP,
             int KP, int NP, const float* __restrict__ bias,
             float* __restrict__ C, int ldc, long c_off, int M, int N)
{
    const int lane = threadIdx.x;
    const int np0 = blockIdx.x * 2;
    const int np1c = (np0 + 1 < NP) ? (np0 + 1) : np0;   // clamped for loads
    const int mp0 = blockIdx.y * 2;

    const v16h* __restrict__ Ap = reinterpret_cast<const v16h*>(AP);
    const v16h* __restrict__ Bp = reinterpret_cast<const v16h*>(WP);
    const v16h* a0p = Ap + (long)mp0 * KP * 32 + lane;         // stride 32/kp
    const v16h* a1p = Ap + (long)(mp0 + 1) * KP * 32 + lane;
    const v16h* b0p = Bp + (long)np0 * 32 + lane;              // stride NP*32/kp
    const v16h* b1p = Bp + (long)np1c * 32 + lane;
    const long bstep = (long)NP * 32;

    v8f acc00 = {}, acc01 = {}, acc10 = {}, acc11 = {};
    for (int kp = 0; kp < KP; ++kp) {
        v16h a0 = a0p[(long)kp * 32];
        v16h a1 = a1p[(long)kp * 32];
        v16h b0 = b0p[(long)kp * bstep];
        v16h b1 = b1p[(long)kp * bstep];
        acc00 = __builtin_amdgcn_wmma_f32_16x16x32_f16(false, a0, false, b0, (short)0, acc00, false, false);
        acc01 = __builtin_amdgcn_wmma_f32_16x16x32_f16(false, a0, false, b1, (short)0, acc01, false, false);
        acc10 = __builtin_amdgcn_wmma_f32_16x16x32_f16(false, a1, false, b0, (short)0, acc10, false, false);
        acc11 = __builtin_amdgcn_wmma_f32_16x16x32_f16(false, a1, false, b1, (short)0, acc11, false, false);
    }

    const int half = lane & 15, hi = lane >> 4;
    const int gn0 = blockIdx.x * 32 + half;
    const int gn1 = gn0 + 16;
    const float bv0 = (bias && gn0 < N) ? bias[gn0] : 0.0f;
    const float bv1 = (bias && gn1 < N) ? bias[gn1] : 0.0f;
    const int m0 = blockIdx.y * 32;
    #pragma unroll
    for (int j = 0; j < 8; ++j) {
        const int gm0 = m0 + hi * 8 + j;   // C layout: VGPR j -> M = j + hi*8
        const int gm1 = gm0 + 16;
        if (gn0 < N) {
            C[c_off + (long)gm0 * ldc + gn0] = acc00[j] + bv0;
            C[c_off + (long)gm1 * ldc + gn0] = acc10[j] + bv0;
        }
        if (gn1 < N) {
            C[c_off + (long)gm0 * ldc + gn1] = acc01[j] + bv1;
            C[c_off + (long)gm1 * ldc + gn1] = acc11[j] + bv1;
        }
    }
}

// ======================= elementwise / scan kernels ========================

// causal depthwise conv (K=4) + SiLU; reads x = xz[:, 0:di]; rev flips time.
// Writes f32 (for the scan) and f16 fragment-packed (for the W_x GEMM).
__global__ void conv_silu_pack(const float* __restrict__ xz, int ldxz,
                               const float* __restrict__ cw, const float* __restrict__ cb,
                               float* __restrict__ xc, _Float16* __restrict__ xcp,
                               int Bb, int L, int di, int rev)
{
    long i = (long)blockIdx.x * blockDim.x + threadIdx.x;
    long total = (long)Bb * L * di;
    if (i >= total) return;
    int d = (int)(i % di);
    long tok = i / di;
    int t = (int)(tok % L);
    long bbase = tok - t;          // b * L
    float acc = cb[d];
    #pragma unroll
    for (int k = 0; k < 4; ++k) {
        int src = rev ? (t + 3 - k) : (t - 3 + k);
        if (src >= 0 && src < L)
            acc += cw[k * di + d] * xz[(bbase + src) * (long)ldxz + d];
    }
    float s = acc / (1.0f + __expf(-acc));   // silu
    xc[tok * (long)di + d] = s;
    xcp[ap_index(tok, d, di >> 5)] = (_Float16)s;   // di % 32 == 0
}

// selective scan; one thread per (b, channel); state h[N] in VGPRs.
// Fused: dt = softplus(dt_raw + b_dt); output y = (scan + x*D) * silu(z),
// written directly as f16 A-fragments for the W_out GEMM.
// Bm/Cm rows are 16B aligned (ldx % 4 == 0, R == 8) -> float4 loads.
template <int N>
__global__ void scan_kernel(const float* __restrict__ xc, const float* __restrict__ dtraw,
                            const float* __restrict__ xdbl, int ldx, int R,
                            const float* __restrict__ A_log, const float* __restrict__ Dp,
                            const float* __restrict__ b_dt,
                            const float* __restrict__ xz, int ldxz,
                            _Float16* __restrict__ yp, int Bb, int L, int di, int rev)
{
    int gid = blockIdx.x * blockDim.x + threadIdx.x;
    if (gid >= Bb * di) return;
    int d = gid % di;
    int b = gid / di;
    float a[N], h[N];
    #pragma unroll
    for (int n = 0; n < N; ++n) {
        a[n] = -__expf(A_log[(long)d * N + n]);
        h[n] = 0.0f;
    }
    float Dd = Dp[d];
    float bd = b_dt[d];
    const int KP = di >> 5;
    for (int tau = 0; tau < L; ++tau) {
        int t = rev ? (L - 1 - tau) : tau;
        long tok = (long)b * L + t;
        float v = dtraw[tok * di + d] + bd;
        float dt_t = (v > 20.0f) ? v : log1pf(__expf(v));   // softplus
        float x_t  = xc[tok * di + d];
        const float4* bc = reinterpret_cast<const float4*>(xdbl + tok * ldx + R);
        float dx = dt_t * x_t;
        float acc = 0.0f;
        #pragma unroll
        for (int q = 0; q < N / 4; ++q) {
            float4 bv = bc[q];          // Bm[4q .. 4q+3]
            float4 cv = bc[N / 4 + q];  // Cm[4q .. 4q+3]
            float h0 = __expf(dt_t * a[4*q+0]) * h[4*q+0] + dx * bv.x;
            float h1 = __expf(dt_t * a[4*q+1]) * h[4*q+1] + dx * bv.y;
            float h2 = __expf(dt_t * a[4*q+2]) * h[4*q+2] + dx * bv.z;
            float h3 = __expf(dt_t * a[4*q+3]) * h[4*q+3] + dx * bv.w;
            h[4*q+0] = h0; h[4*q+1] = h1; h[4*q+2] = h2; h[4*q+3] = h3;
            acc += h0 * cv.x + h1 * cv.y + h2 * cv.z + h3 * cv.w;
        }
        float yv = acc + x_t * Dd;
        float z = xz[tok * (long)ldxz + di + d];
        yv *= z / (1.0f + __expf(-z));                      // * silu(z)
        yp[ap_index(tok, d, KP)] = (_Float16)yv;
    }
}

// x = LN(x + add) * g + beta   (biased var), one wave per token, HD=128
__global__ __launch_bounds__(32)
void residual_ln(float* __restrict__ x, const float* __restrict__ add,
                 const float* __restrict__ g, const float* __restrict__ beta, int T)
{
    int tok = blockIdx.x;
    int lane = threadIdx.x;
    long base = (long)tok * HD;
    float v[4];
    float s = 0.0f;
    #pragma unroll
    for (int i = 0; i < 4; ++i) {
        v[i] = x[base + lane + 32 * i] + add[base + lane + 32 * i];
        s += v[i];
    }
    #pragma unroll
    for (int o = 16; o; o >>= 1) s += __shfl_xor(s, o, 32);
    float mean = s * (1.0f / HD);
    float vs = 0.0f;
    #pragma unroll
    for (int i = 0; i < 4; ++i) { float dd = v[i] - mean; vs += dd * dd; }
    #pragma unroll
    for (int o = 16; o; o >>= 1) vs += __shfl_xor(vs, o, 32);
    float inv = rsqrtf(vs * (1.0f / HD) + 1e-5f);
    #pragma unroll
    for (int i = 0; i < 4; ++i)
        x[base + lane + 32 * i] =
            (v[i] - mean) * inv * g[lane + 32 * i] + beta[lane + 32 * i];
}

// per-token variance, ddof=1
__global__ __launch_bounds__(32)
void var_token(const float* __restrict__ z, float* __restrict__ var, int T)
{
    int tok = blockIdx.x;
    int lane = threadIdx.x;
    long base = (long)tok * HD;
    float v[4];
    float s = 0.0f;
    #pragma unroll
    for (int i = 0; i < 4; ++i) { v[i] = z[base + lane + 32 * i]; s += v[i]; }
    #pragma unroll
    for (int o = 16; o; o >>= 1) s += __shfl_xor(s, o, 32);
    float mean = s * (1.0f / HD);
    float vs = 0.0f;
    #pragma unroll
    for (int i = 0; i < 4; ++i) { float dd = v[i] - mean; vs += dd * dd; }
    #pragma unroll
    for (int o = 16; o; o >>= 1) vs += __shfl_xor(vs, o, 32);
    if (lane == 0) var[tok] = vs * (1.0f / (HD - 1));
}

__global__ __launch_bounds__(256)
void max_seq(const float* __restrict__ var, float* __restrict__ vmax, int L)
{
    __shared__ float sm[256];
    int b = blockIdx.x;
    float m = -3.4e38f;
    for (int t = threadIdx.x; t < L; t += 256) m = fmaxf(m, var[(long)b * L + t]);
    sm[threadIdx.x] = m;
    __syncthreads();
    for (int o = 128; o; o >>= 1) {
        if ((int)threadIdx.x < o) sm[threadIdx.x] = fmaxf(sm[threadIdx.x], sm[threadIdx.x + o]);
        __syncthreads();
    }
    if (threadIdx.x == 0) vmax[b] = sm[0];
}

// z = z * sigmoid(lin) * (1 + var/(vmax+1e-6))
__global__ void apply_gate(float* __restrict__ z, const float* __restrict__ lin,
                           const float* __restrict__ var, const float* __restrict__ vmax,
                           int L, long T)
{
    long i = (long)blockIdx.x * blockDim.x + threadIdx.x;
    if (i >= T * HD) return;
    long tok = i / HD;
    int b = (int)(tok / L);
    float gsig = 1.0f / (1.0f + __expf(-lin[i]));
    float vn = var[tok] / (vmax[b] + 1e-6f);
    z[i] = z[i] * gsig * (1.0f + vn);
}

// x_seq = concat([z_t, z_a, z_v], axis=1)  ->  (4, 480, 128)
__global__ void concat_seq(const float* __restrict__ zt, const float* __restrict__ za,
                           const float* __restrict__ zv, float* __restrict__ xseq)
{
    long i = (long)blockIdx.x * blockDim.x + threadIdx.x;
    if (i >= (long)4 * 480 * HD) return;
    int d = (int)(i % HD);
    long tok = i / HD;
    int t = (int)(tok % 480);
    int b = (int)(tok / 480);
    const float* src; long r;
    if (t < 160)      { src = zt; r = (long)b * 160 + t; }
    else if (t < 360) { src = za; r = (long)b * 200 + (t - 160); }
    else              { src = zv; r = (long)b * 120 + (t - 360); }
    xseq[i] = src[r * HD + d];
}

// xm[(b*200+tau)*3 + m] = xseq[b, off_m + tau]  (zero-padded past L_m)
__global__ void build_xm(const float* __restrict__ xseq, float* __restrict__ xm)
{
    long i = (long)blockIdx.x * blockDim.x + threadIdx.x;
    if (i >= (long)4 * 200 * 3 * HD) return;
    int d = (int)(i % HD);
    long rem = i / HD;
    int m = (int)(rem % 3); rem /= 3;
    int tau = (int)(rem % 200);
    int b = (int)(rem / 200);
    int Lm  = (m == 0) ? 160 : (m == 1 ? 200 : 120);
    int off = (m == 0) ? 0   : (m == 1 ? 160 : 360);
    float v = 0.0f;
    if (tau < Lm) v = xseq[((long)b * 480 + off + tau) * HD + d];
    xm[i] = v;
}

// scatter modality-mamba output back to sequence layout in concat buffer
__global__ void scatter_hm(const float* __restrict__ out, float* __restrict__ cat, int col)
{
    long i = (long)blockIdx.x * blockDim.x + threadIdx.x;
    if (i >= (long)4 * 480 * HD) return;
    int d = (int)(i % HD);
    long tok = i / HD;
    int t = (int)(tok % 480);
    int b = (int)(tok / 480);
    int m, tau;
    if (t < 160)      { m = 0; tau = t; }
    else if (t < 360) { m = 1; tau = t - 160; }
    else              { m = 2; tau = t - 360; }
    cat[tok * 512 + col + d] = out[(((long)b * 200 + tau) * 3 + m) * HD + d];
}

// ======================= host-side orchestration ===========================

struct MambaP {
    const float *W_in, *conv_w, *conv_b, *W_x, *W_dt, *b_dt, *A_log, *D, *W_out;
};
static inline const float* F(void* const* din, int i) { return (const float*)din[i]; }
static inline MambaP mk(void* const* din, int b) {
    return { F(din,b), F(din,b+1), F(din,b+2), F(din,b+3), F(din,b+4),
             F(din,b+5), F(din,b+6), F(din,b+7), F(din,b+8) };
}

struct Scratch {
    float *XZ, *XC, *XDBL, *DT, *OUT;
    _Float16 *XCP, *YP, *APK, *WPK;
};

// pack W + WMMA GEMM with pre-packed A  (M must be a multiple of 32)
static void gemm_pA(const _Float16* AP, int KP, int M,
                    const float* W, int ldw, int K, int N, const float* bias,
                    float* C, int ldc, long c_off, const Scratch& s, hipStream_t st)
{
    const int NP = (N + 15) / 16;
    const long wtot = (long)KP * NP * 512;
    pack_w<<<(int)((wtot + 255) / 256), 256, 0, st>>>(W, ldw, K, N, s.WPK, KP, NP);
    gemm_pk<<<dim3((NP + 1) / 2, M / 32), 32, 0, st>>>(AP, s.WPK, KP, NP,
                                                       bias, C, ldc, c_off, M, N);
}

// pack A + pack W + WMMA GEMM
static void gemm(const float* A, int lda, long a_off, int M, int K,
                 const float* W, int ldw, int N, const float* bias,
                 float* C, int ldc, long c_off, const Scratch& s, hipStream_t st)
{
    const int KP = (K + 31) / 32;
    const long atot = (long)(M / 16) * KP * 512;
    pack_a<<<(int)((atot + 255) / 256), 256, 0, st>>>(A, lda, a_off, M, K, s.APK, KP);
    gemm_pA(s.APK, KP, M, W, ldw, K, N, bias, C, ldc, c_off, s, st);
}

static void run_mamba(const MambaP& p, const float* x_in, float* outC, int ldc, long c_off,
                      int Bb, int L, int di, int N, int rev,
                      const Scratch& w, hipStream_t s)
{
    const int R = 8;
    const long T = (long)Bb * L;      // always a multiple of 32 here
    const int ldxz = 2 * di;
    const int ldx  = R + 2 * N;
    const long tot = T * di;

    // xz = x @ W_in
    gemm(x_in, HD, 0, (int)T, HD, p.W_in, 2 * di, 2 * di, nullptr,
         w.XZ, ldxz, 0, w, s);
    // causal conv + silu -> XC (f32) and XCP (f16 fragments)
    conv_silu_pack<<<(int)((tot + 255) / 256), 256, 0, s>>>(
        w.XZ, ldxz, p.conv_w, p.conv_b, w.XC, w.XCP, Bb, L, di, rev);
    // x_dbl = xc @ W_x   (A already packed by conv)
    gemm_pA(w.XCP, di / 32, (int)T, p.W_x, ldx, di, ldx, nullptr,
            w.XDBL, ldx, 0, w, s);
    // dt_raw = x_dbl[:, :R] @ W_dt   (softplus + b_dt fused into the scan)
    gemm(w.XDBL, ldx, 0, (int)T, R, p.W_dt, di, di, nullptr,
         w.DT, di, 0, w, s);
    // selective scan (+ D skip, softplus, silu(z) gate) -> YP (f16 fragments)
    int nthr = Bb * di;
    if (N == 32)
        scan_kernel<32><<<(nthr + 127) / 128, 128, 0, s>>>(
            w.XC, w.DT, w.XDBL, ldx, R, p.A_log, p.D, p.b_dt,
            w.XZ, ldxz, w.YP, Bb, L, di, rev);
    else
        scan_kernel<64><<<(nthr + 127) / 128, 128, 0, s>>>(
            w.XC, w.DT, w.XDBL, ldx, R, p.A_log, p.D, p.b_dt,
            w.XZ, ldxz, w.YP, Bb, L, di, rev);
    // out = (y * silu(z)) @ W_out   (A already packed by scan)
    gemm_pA(w.YP, di / 32, (int)T, p.W_out, HD, di, HD, nullptr,
            outC, ldc, c_off, w, s);
}

extern "C" void kernel_launch(void* const* d_in, const int* in_sizes, int n_in,
                              void* d_out, int out_size, void* d_ws, size_t ws_size,
                              hipStream_t stream)
{
    (void)in_sizes; (void)n_in; (void)out_size; (void)ws_size;
    const int B = 4, LT = 160, LA = 200, LV = 120, LS = 480;

    // --- workspace layout (floats; every offset a multiple of 8 -> 32B) ---
    float* w = (float*)d_ws;
    float* XSEQ = w;                 // 4*480*128      = 245760
    float* ZT   = XSEQ + 245760;     // 640*128        = 81920
    float* ZA   = ZT   + 81920;      // 800*128        = 102400
    float* ZV   = ZA   + 102400;     // 480*128        = 61440
    float* XZ   = ZV   + 61440;      // 2400*1024      = 2457600
    float* XC   = XZ   + 2457600;    // 2400*512       = 1228800
    float* XDBL = XC   + 1228800;    // 2400*136       = 326400
    float* DT   = XDBL + 326400;     // 2400*512       = 1228800
    float* XCPf = DT   + 1228800;    // 1228800 f16    = 614400 floats
    float* YPf  = XCPf + 614400;     // 1228800 f16    = 614400 floats
    float* OUT  = YPf  + 614400;     // 2400*128       = 307200
    float* CAT  = OUT  + 307200;     // 1920*512       = 983040
    float* XM   = CAT  + 983040;     // 2400*128       = 307200
    float* GLIN = XM   + 307200;     // 800*128        = 102400
    float* APKf = GLIN + 102400;     // 983040 f16     = 491520 floats
    float* WPKf = APKf + 491520;     // 131072 f16     = 65536 floats
    float* VAR  = WPKf + 65536;      // 800
    float* VMAX = VAR  + 800;        // 8
    Scratch sc = { XZ, XC, XDBL, DT, OUT,
                   (_Float16*)XCPf, (_Float16*)YPf,
                   (_Float16*)APKf, (_Float16*)WPKf };

    // --- input index map (setup_inputs dict-insertion order) ---
    // 0..5: text, audio, vision, text_mask, audio_mask, vision_mask
    // 6,7: proj_t.W,b  8,9: proj_a  10,11: proj_v
    // 12+11l: intra_t layer (W_in..W_out, g, beta); 45: intra_a; 78: intra_v
    // 111..116: gate_t/a/v (W,b)   117..119: mwg (dead code)
    // 120+40f: fusion block (t_fwd 0..8, t_bwd 9..17, m_fwd 18..26,
    //                        m_bwd 27..35, Wp 36, bp 37, g 38, beta 39)
    const float* text   = F(d_in, 0);
    const float* audio  = F(d_in, 1);
    const float* vision = F(d_in, 2);

    // --- input projections ---
    gemm(text,   768, 0, B * LT, 768, F(d_in, 6),  HD, HD, F(d_in, 7),
         ZT, HD, 0, sc, stream);
    gemm(audio,  74,  0, B * LA, 74,  F(d_in, 8),  HD, HD, F(d_in, 9),
         ZA, HD, 0, sc, stream);
    gemm(vision, 35,  0, B * LV, 35,  F(d_in, 10), HD, HD, F(d_in, 11),
         ZV, HD, 0, sc, stream);

    // --- intra-modal stacks (di=256, N=32) + residual LN ---
    struct { float* z; int L; int base; } intra[3] = {
        { ZT, LT, 12 }, { ZA, LA, 45 }, { ZV, LV, 78 } };
    for (int m = 0; m < 3; ++m) {
        for (int l = 0; l < 3; ++l) {
            int ib = intra[m].base + l * 11;
            run_mamba(mk(d_in, ib), intra[m].z, OUT, HD, 0,
                      B, intra[m].L, 256, 32, 0, sc, stream);
            residual_ln<<<B * intra[m].L, 32, 0, stream>>>(
                intra[m].z, OUT, F(d_in, ib + 9), F(d_in, ib + 10), B * intra[m].L);
        }
    }

    // --- align gates (variance-scaled sigmoid gate) ---
    int gidx[3] = { 111, 113, 115 };
    for (int m = 0; m < 3; ++m) {
        long T = (long)B * intra[m].L;
        gemm(intra[m].z, HD, 0, (int)T, HD, F(d_in, gidx[m]), HD, HD,
             F(d_in, gidx[m] + 1), GLIN, HD, 0, sc, stream);
        var_token<<<(int)T, 32, 0, stream>>>(intra[m].z, VAR, (int)T);
        max_seq<<<B, 256, 0, stream>>>(VAR, VMAX, intra[m].L);
        apply_gate<<<(int)((T * HD + 255) / 256), 256, 0, stream>>>(
            intra[m].z, GLIN, VAR, VMAX, intra[m].L, T);
    }

    // modality weight gating: softmax over a size-1 axis => weights == 1.0
    // (masked_mean / mwg are dead code; skipped.)

    // --- fusion: x_seq = concat(z_t, z_a, z_v) ---
    long nseq = (long)B * LS * HD;
    concat_seq<<<(int)((nseq + 255) / 256), 256, 0, stream>>>(ZT, ZA, ZV, XSEQ);

    for (int f = 0; f < 2; ++f) {
        int fb = 120 + f * 40;
        // temporal fwd/bwd mambas -> concat buffer cols [0:128) and [128:256)
        run_mamba(mk(d_in, fb + 0), XSEQ, CAT, 512, 0,   B, LS, 512, 64, 0, sc, stream);
        run_mamba(mk(d_in, fb + 9), XSEQ, CAT, 512, 128, B, LS, 512, 64, 1, sc, stream);
        // modality-axis mambas over xm (B*Lmax=800, L=3)
        build_xm<<<(int)(((long)4 * 200 * 3 * HD + 255) / 256), 256, 0, stream>>>(XSEQ, XM);
        run_mamba(mk(d_in, fb + 18), XM, OUT, HD, 0, 800, 3, 512, 64, 0, sc, stream);
        scatter_hm<<<(int)((nseq + 255) / 256), 256, 0, stream>>>(OUT, CAT, 256);
        run_mamba(mk(d_in, fb + 27), XM, OUT, HD, 0, 800, 3, 512, 64, 1, sc, stream);
        scatter_hm<<<(int)((nseq + 255) / 256), 256, 0, stream>>>(OUT, CAT, 384);
        // fused = cat @ Wp + bp ; x_seq = LN(fused + x_seq)
        gemm(CAT, 512, 0, B * LS, 512, F(d_in, fb + 36), HD, HD,
             F(d_in, fb + 37), OUT, HD, 0, sc, stream);
        residual_ln<<<B * LS, 32, 0, stream>>>(
            XSEQ, OUT, F(d_in, fb + 38), F(d_in, fb + 39), B * LS);
    }

    hipMemcpyAsync(d_out, XSEQ, (size_t)nseq * sizeof(float),
                   hipMemcpyDeviceToDevice, stream);
}
